// TokenQKV_22462678958126
// MI455X (gfx1250) — matrix operand order repaired and checked by reading, not compiled, not run on hardware
//
#include <hip/hip_runtime.h>
#include <hip/hip_bf16.h>

typedef __attribute__((ext_vector_type(16))) _Float16 v16h;
typedef __attribute__((ext_vector_type(8)))  float    v8f;
typedef __attribute__((ext_vector_type(4)))  unsigned int u32x4;
typedef __attribute__((ext_vector_type(8)))  int      i32x8;
typedef __attribute__((ext_vector_type(4)))  int      i32x4;

#define EMBED 384
#define HEADS 6
#define HDIM  64
#define HW    196
#define WPAD  224   // 7 * 32, exact K-dim tiling for PV
#define MSUP  25
#define MPAD  32
#define PIMG  75
#define NBATCH 8
#define QSCALE 0.125f   // 64^-0.5

// ---- Tensor Data Mover: 2D f32 tile, global -> LDS (ISA 08_async_tensor §8) ----
// D# group0: count=1 | lds_addr | global_addr[56:0] | type=2 ("image")
// D# group1: data_size=4B, tensor_dim0/1, tile_dim0/1, tensor_dim0_stride
__device__ __forceinline__ void tdm_load_2d_f32(const float* gsrc, unsigned lds_byte_off,
                                                int tile_w, int tile_h, int row_stride) {
  unsigned long long ga = (unsigned long long)(size_t)gsrc;
  u32x4 g0;
  g0[0] = 1u;                                        // count=1 (valid user D#)
  g0[1] = lds_byte_off;                              // lds_addr (bytes)
  g0[2] = (unsigned)(ga & 0xFFFFFFFFu);              // global_addr[31:0]
  g0[3] = ((unsigned)(ga >> 32) & 0x01FFFFFFu)       // global_addr[56:32]
        | (2u << 30);                                // type=2
  i32x8 g1;
  g1[0] = (2 << 16);                                 // data_size=2 -> 4 bytes; wg_mask=0
  g1[1] = (tile_w & 0xFFFF) << 16;                   // tensor_dim0[15:0]
  g1[2] = ((tile_w >> 16) & 0xFFFF)                  // tensor_dim0[31:16]
        | ((tile_h & 0xFFFF) << 16);                 // tensor_dim1[15:0]
  g1[3] = ((tile_h >> 16) & 0xFFFF)                  // tensor_dim1[31:16]
        | ((tile_w & 0xFFFF) << 16);                 // tile_dim0
  g1[4] = (tile_h & 0xFFFF);                         // tile_dim1 ; tile_dim2=0 (unused)
  g1[5] = row_stride;                                // tensor_dim0_stride[31:0]
  g1[6] = 0;                                         // stride hi / tensor_dim1_stride
  g1[7] = 0;
  i32x4 gz = {0, 0, 0, 0};
#if defined(__clang_major__) && (__clang_major__ >= 23)
  i32x8 gz8 = {0, 0, 0, 0, 0, 0, 0, 0};
  __builtin_amdgcn_tensor_load_to_lds(g0, g1, gz, gz, gz8, 0);
#else
  __builtin_amdgcn_tensor_load_to_lds(g0, g1, gz, gz, 0);
#endif
}

// ---- fragment loaders (16-bit WMMA operands; contiguous k = 16*hi + e, same in A and B) ----

// A-matrix 16x32 f16: lane(0..15)=row M, elems 0..15 -> K=(0..15)+16*hi
__device__ __forceinline__ v16h frag_a_rowmajor(const _Float16* X, int pitch,
                                                int row_base, int k_base, int lane) {
  const int m  = lane & 15;
  const int hi = lane >> 4;
  const _Float16* p = X + (row_base + m) * pitch + k_base + 16 * hi;
  v16h r;
#pragma unroll
  for (int e = 0; e < 8; ++e) r[e] = p[e];
#pragma unroll
  for (int e = 0; e < 8; ++e) r[8 + e] = p[8 + e];
  return r;
}

// B-matrix 32x16 f16: lane holds column n=lane&15, K = e + 16*(lane>=16).
// Here B[k][n] = X[n_base + n][k_base + k]  (X row-major with rows = N dim, i.e. B = X^T)
__device__ __forceinline__ v16h frag_b_rowsN(const _Float16* X, int pitch,
                                             int n_base, int k_base, int lane) {
  const int n  = lane & 15;
  const int hi = lane >> 4;
  const _Float16* p = X + (n_base + n) * pitch + k_base + 16 * hi;
  v16h r;
#pragma unroll
  for (int e = 0; e < 16; ++e) r[e] = p[e];
  return r;
}

// B[k][n] = X[k_base + k][n_base + n]  (X row-major with rows = K dim) -> column gather
__device__ __forceinline__ v16h frag_b_rowsK(const _Float16* X, int pitch,
                                             int k_base, int n_base, int lane) {
  const int n  = lane & 15;
  const int hi = lane >> 4;
  const _Float16* p = X + (k_base + 16 * hi) * pitch + n_base + n;
  v16h r;
#pragma unroll
  for (int e = 0; e < 16; ++e) r[e] = p[e * pitch];
  return r;
}

// ---- main kernel: one workgroup per (b, p); 8 wave32s ----
__global__ void __launch_bounds__(256)
fewshot_attn_kernel(const float* __restrict__ xq,   // (8, 75, 196, 384)
                    const float* __restrict__ xs,   // (8, 25, 384)
                    float* __restrict__ out) {      // (8, 75, 25, 384)
  __shared__ __align__(16) float    sKV32[HW * HDIM];   // 49 KB TDM staging (f32)
  __shared__ _Float16 sQ [MPAD * HDIM];                 //  4 KB
  __shared__ _Float16 sKV[WPAD * HDIM];                 // 28 KB
  __shared__ _Float16 sS [MPAD * WPAD];                 // 14 KB (logits -> attn in place)

  const int bp   = blockIdx.x;            // 0..599
  const int b    = bp / PIMG;
  const int p    = bp % PIMG;
  const int tid  = threadIdx.x;
  const int lane = tid & 31;
  const int wave = tid >> 5;

  const float* kv_base = xq + (size_t)(b * PIMG + p) * HW * EMBED;
  const float* q_base  = xs + (size_t)b * MSUP * EMBED;
  float*       o_base  = out + (size_t)(b * PIMG + p) * MSUP * EMBED;

  const unsigned stage_lds = (unsigned)(size_t)(void*)sKV32;  // low 32 bits = LDS offset

  // Prologue: kick off DMA of head 0's KV slice (one TDM per workgroup; EXEC-independent)
  if (wave == 0)
    tdm_load_2d_f32(kv_base + 0 * HDIM, stage_lds, HDIM, HW, EMBED);

  for (int h = 0; h < HEADS; ++h) {
    // Wait for this head's tensor DMA (wave 0 owns TENSORcnt; barrier publishes to all)
    __builtin_amdgcn_s_wait_tensorcnt(0);
    __syncthreads();

    // ---- Phase A: convert staged KV f32 -> f16 (zero-pad w >= 196); stage Q ----
    for (int i = tid; i < WPAD * HDIM; i += 256) {
      const int w = i >> 6, d = i & 63;
      const float v = (w < HW) ? sKV32[w * HDIM + d] : 0.f;
      sKV[i] = (_Float16)v;
    }
    for (int i = tid; i < MPAD * HDIM; i += 256) {
      const int r = i >> 6, d = i & 63;
      const float v = (r < MSUP) ? q_base[r * EMBED + h * HDIM + d] : 0.f;
      sQ[i] = (_Float16)v;
    }
    __syncthreads();

    // Staging buffer is free now: prefetch next head's KV via TDM, overlapped with compute
    if ((h + 1) < HEADS && wave == 0)
      tdm_load_2d_f32(kv_base + (h + 1) * HDIM, stage_lds, HDIM, HW, EMBED);

    // ---- Phase B: S = Q * K^T  (2 m-tiles x 14 w-tiles, K-dim 64 = 2 wmma) ----
    for (int j = wave; j < 2 * (WPAD / 16); j += 8) {
      const int mt = j / (WPAD / 16);
      const int wt = j % (WPAD / 16);
      const int m0 = mt * 16, w0 = wt * 16;
      v8f acc = {};
#pragma unroll
      for (int kc = 0; kc < HDIM; kc += 32) {
        v16h a  = frag_a_rowmajor(sQ,  HDIM, m0, kc, lane);
        v16h bm = frag_b_rowsN   (sKV, HDIM, w0, kc, lane);   // B = K^T
        acc = __builtin_amdgcn_wmma_f32_16x16x32_f16(false, a, false, bm,
                                                     (short)0, acc, false, false);
      }
      const int hi = lane >> 4, n = lane & 15;
#pragma unroll
      for (int r = 0; r < 8; ++r)
        sS[(m0 + r + 8 * hi) * WPAD + w0 + n] = (_Float16)(acc[r] * QSCALE);
    }
    __syncthreads();

    // ---- Phase C: per-row softmax over valid w (0..195), in-place f16 ----
    if (tid < MPAD) {
      const int r = tid;
      float mx = -3.0e38f;
      for (int c = 0; c < HW; ++c) mx = fmaxf(mx, (float)sS[r * WPAD + c]);
      float sum = 0.f;
      for (int c = 0; c < HW; ++c) sum += __expf((float)sS[r * WPAD + c] - mx);
      const float inv = 1.f / sum;
      for (int c = 0; c < WPAD; ++c) {
        const float v = (c < HW) ? __expf((float)sS[r * WPAD + c] - mx) * inv : 0.f;
        sS[r * WPAD + c] = (_Float16)v;   // same-element rewrite: race-free
      }
    }
    __syncthreads();

    // ---- Phase D: O = A * V  (2 m-tiles x 4 d-tiles = 8 jobs, one per wave) ----
    {
      const int mt = wave >> 2, dt = wave & 3;
      const int m0 = mt * 16, d0 = dt * 16;
      v8f acc = {};
#pragma unroll
      for (int kc = 0; kc < WPAD; kc += 32) {
        v16h a  = frag_a_rowmajor(sS,  WPAD, m0, kc, lane);
        v16h bm = frag_b_rowsK   (sKV, HDIM, kc, d0, lane);   // B = V
        acc = __builtin_amdgcn_wmma_f32_16x16x32_f16(false, a, false, bm,
                                                     (short)0, acc, false, false);
      }
      const int hi = lane >> 4, n = lane & 15;
#pragma unroll
      for (int r = 0; r < 8; ++r) {
        const int m = m0 + r + 8 * hi;
        if (m < MSUP)
          o_base[m * EMBED + h * HDIM + d0 + n] = acc[r];
      }
    }
    // next iteration's leading s_wait_tensorcnt + barrier protects LDS reuse
  }
}

// ---- o_support passthrough: (8, 1, 25, 384) = x_support flat copy ----
__global__ void __launch_bounds__(256)
support_copy_kernel(const float* __restrict__ xs, float* __restrict__ out) {
  const int i = blockIdx.x * 256 + threadIdx.x;
  if (i < NBATCH * MSUP * EMBED) out[i] = xs[i];
}

extern "C" void kernel_launch(void* const* d_in, const int* in_sizes, int n_in,
                              void* d_out, int out_size, void* d_ws, size_t ws_size,
                              hipStream_t stream) {
  const float* xq = (const float*)d_in[0];   // x_query  (8,5,15,196,384)
  const float* xs = (const float*)d_in[1];   // x_support(8,5,5,384)
  float* out = (float*)d_out;

  const int oq_elems = NBATCH * PIMG * MSUP * EMBED;   // 5,760,000
  const int os_elems = NBATCH * MSUP * EMBED;          //    76,800

  fewshot_attn_kernel<<<NBATCH * PIMG, 256, 0, stream>>>(xq, xs, out);
  support_copy_kernel<<<(os_elems + 255) / 256, 256, 0, stream>>>(xs, out + oq_elems);
  (void)oq_elems; (void)in_sizes; (void)n_in; (void)out_size; (void)d_ws; (void)ws_size;
}
